// Spatial_PDE_SSM_Block_41635412967881
// MI455X (gfx1250) — matrix operand order repaired and checked by reading, not compiled, not run on hardware
//
#include <hip/hip_runtime.h>
#include <hip/hip_bf16.h>
#include <stdint.h>

// ---------------------------------------------------------------------------
// CDNA5 / gfx1250 (MI455X) implementation of the Spatial-PDE-SSM block.
// wave32; all channel-mixing 1x1 convs + windowed attention use
// v_wmma_f32_16x16x32_bf16 (f32 accumulate). The memory-bound hidden-state
// update (206 MB of h traffic -> ~9us HBM floor @ 23.3 TB/s) is one fused
// coalesced pass. Activations stream as bf16 between GEMMs to halve bytes.
// New this round: GEMM W-tiles are staged into LDS by the Tensor Data Mover
// (tensor_load_to_lds, TENSORcnt) instead of wave VMEM loads.
// ---------------------------------------------------------------------------

typedef __attribute__((ext_vector_type(16))) __bf16 v16bf;
typedef __attribute__((ext_vector_type(8)))  __bf16 v8bf;
typedef __attribute__((ext_vector_type(8)))  float  v8f;
typedef __attribute__((ext_vector_type(4)))  unsigned int u32x4;
typedef __attribute__((ext_vector_type(8)))  int i32x8;
typedef __attribute__((ext_vector_type(4)))  int i32x4;

#define USE_TDM 1

#define DEV static __device__ __forceinline__

DEV __bf16 tobf(float f) { return (__bf16)f; }
DEV float sigf(float v)  { return 1.0f / (1.0f + __expf(-v)); }

// Build a 16-bit A fragment (16x32, MxK) in contiguous k order (not the ISA table order):
// lane l (row = l&15): kb = (l>>4)*16 ; elements = K[kb..kb+15] (matches B)
DEV v16bf load_frag_a(const __bf16* rowp, int kb2) {
  v8bf lo = *(const v8bf*)(rowp + 2 * kb2);
  v8bf hi = *(const v8bf*)(rowp + 2 * kb2 + 8);
  return __builtin_shufflevector(lo, hi, 0,1,2,3,4,5,6,7,8,9,10,11,12,13,14,15);
}
// B fragment (32x16, KxN): lane l (col = l&15): K = (l>>4)*16 + 0..15,
// i.e. 16 contiguous K values -> one 32B LDS read from a [N][K] layout.

#if USE_TDM
// Issue a TDM 2D tile load: rows x cols bf16 elements, global row pitch
// `pitch` elements, into LDS at `lds_byte_off` (row-major, tightly packed).
// D# layout per CDNA5 ISA (group0/group1).
DEV void tdm_load_tile_bf16(const void* gptr, unsigned lds_byte_off,
                            int rows, int cols, int pitch) {
  unsigned long long ga = (unsigned long long)(uintptr_t)gptr;
  u32x4 g0;
  g0[0] = 1u;                                   // count=1, user descriptor
  g0[1] = lds_byte_off;                         // lds_addr (bytes)
  g0[2] = (unsigned)(ga & 0xFFFFFFFFu);         // global_addr[31:0]
  g0[3] = (unsigned)((ga >> 32) & 0x01FFFFFFu)  // global_addr[56:32]
          | 0x80000000u;                        // type=2 ("image")
  i32x8 g1;
  g1[0] = 0x00010000;                           // data_size=1 (2 bytes)
  g1[1] = (int)(((unsigned)cols & 0xFFFFu) << 16);        // tensor_dim0 lo
  g1[2] = (int)((((unsigned)cols >> 16) & 0xFFFFu)
          | (((unsigned)rows & 0xFFFFu) << 16));          // d0 hi | d1 lo
  g1[3] = (int)(((unsigned)cols & 0xFFFFu) << 16);        // d1 hi=0 | tile_dim0
  g1[4] = rows;                                 // tile_dim1 | tile_dim2=0
  g1[5] = pitch;                                // tensor_dim0_stride lo32
  g1[6] = 0;                                    // stride hi | d1_stride lo
  g1[7] = 0;
  i32x4 gz = {0, 0, 0, 0};
#if __clang_major__ >= 23
  i32x8 gz8 = {0, 0, 0, 0, 0, 0, 0, 0};
  __builtin_amdgcn_tensor_load_to_lds(g0, g1, gz, gz, gz8, 0);
#else
  __builtin_amdgcn_tensor_load_to_lds(g0, g1, gz, gz, 0);
#endif
}
#endif

// ---------------------------------------------------------------------------
// Generic WMMA GEMM: Y[M x 4096] = W[M x Kp] @ X[Kp x 4096] (+bias, act,
// +residual, dual f32/bf16 store). Block = 256 threads = 8 waves,
// tile 64(M) x 128(N), K-step 32. ACT: 0=none, 1=sigmoid.
// W tile (64x32, row-major) is DMA'd by the TDM; X tile is transposed into
// LDS by the waves (TDM cannot transpose).
// ---------------------------------------------------------------------------
template <int ACT>
__global__ __launch_bounds__(256)
void gemm_wmma_kernel(const __bf16* __restrict__ Wm, const __bf16* __restrict__ Xm,
                      const float* __restrict__ bias, const float* __restrict__ resid,
                      float* __restrict__ outF, __bf16* __restrict__ outB,
                      int M, int Kp) {
  __shared__ __bf16 Wl[64][32];    // [m][k]  (TDM destination, row-major)
  __shared__ __bf16 Xl[128][32];   // transposed: [n][k]
  const int n0 = blockIdx.x * 128;
  const int m0 = blockIdx.y * 64;
  const int tid  = threadIdx.x;
  const int lane = tid & 31, wave = tid >> 5;
  const int wm = wave & 1, wn = wave >> 1;        // 2 x 4 waves of 32x32
  v8f acc00 = {}, acc01 = {}, acc10 = {}, acc11 = {};
  const int xk = tid >> 3,  xnc = (tid & 7) * 16; // X staging: 32 k x 128 n
#if !USE_TDM
  const int wrow = tid >> 2, wkc = (tid & 3) * 8; // W staging: 64 rows x 32
#endif

  for (int kb = 0; kb < Kp; kb += 32) {
#if USE_TDM
    // stage W tile via the Tensor Data Mover (wave 0 issues; EXEC ignored)
    if (wave == 0) {
      tdm_load_tile_bf16(Wm + (size_t)m0 * Kp + kb,
                         (unsigned)(uintptr_t)&Wl[0][0], 64, 32, Kp);
    }
#else
    *(uint4*)&Wl[wrow][wkc] =
        *(const uint4*)(Wm + (size_t)(m0 + wrow) * Kp + kb + wkc);
#endif
    // stage X tile, transposing [k][n] -> LDS [n][k]
    union { uint4 q[2]; __bf16 h[16]; } xt;
    const uint4* gx = (const uint4*)(Xm + (size_t)(kb + xk) * 4096 + n0 + xnc);
    xt.q[0] = gx[0]; xt.q[1] = gx[1];
#pragma unroll
    for (int j = 0; j < 16; ++j) Xl[xnc + j][xk] = xt.h[j];
    if (kb + 32 < Kp)  // pull next K-slab toward L2/L0 (global_prefetch_b8)
      __builtin_prefetch(Xm + (size_t)(kb + 32 + xk) * 4096 + n0 + xnc, 0, 0);
#if USE_TDM
    if (wave == 0) __builtin_amdgcn_s_wait_tensorcnt(0);  // W tile landed
#endif
    __syncthreads();

    const int kb2 = (lane >> 4) * 8;
    const int kh  = (lane >> 4) * 16;
    v16bf a0 = load_frag_a(&Wl[wm * 32 +      (lane & 15)][0], kb2);
    v16bf a1 = load_frag_a(&Wl[wm * 32 + 16 + (lane & 15)][0], kb2);
    v16bf b0 = *(const v16bf*)&Xl[wn * 32 +      (lane & 15)][kh];
    v16bf b1 = *(const v16bf*)&Xl[wn * 32 + 16 + (lane & 15)][kh];
    acc00 = __builtin_amdgcn_wmma_f32_16x16x32_bf16(false, a0, false, b0, (short)0, acc00, false, false);
    acc01 = __builtin_amdgcn_wmma_f32_16x16x32_bf16(false, a0, false, b1, (short)0, acc01, false, false);
    acc10 = __builtin_amdgcn_wmma_f32_16x16x32_bf16(false, a1, false, b0, (short)0, acc10, false, false);
    acc11 = __builtin_amdgcn_wmma_f32_16x16x32_bf16(false, a1, false, b1, (short)0, acc11, false, false);
    __syncthreads();  // also keeps next TDM from overwriting live Wl
  }

  // epilogue: D layout (7.12.2): lane -> col = lane&15, rows = vg + 8*(lane>>4)
  const int ml = lane >> 4, nl = lane & 15;
  v8f accs[2][2] = {{acc00, acc01}, {acc10, acc11}};
#pragma unroll
  for (int r = 0; r < 2; ++r)
#pragma unroll
    for (int c2 = 0; c2 < 2; ++c2)
#pragma unroll
      for (int vg = 0; vg < 8; ++vg) {
        int m = m0 + wm * 32 + r * 16 + vg + 8 * ml;
        int n = n0 + wn * 32 + c2 * 16 + nl;
        if (m < M) {
          float v = accs[r][c2][vg];
          if (bias)  v += bias[m];
          if (ACT == 1) v = sigf(v);
          if (resid) v += resid[(size_t)m * 4096 + n];
          if (outF)  outF[(size_t)m * 4096 + n] = v;
          if (outB)  outB[(size_t)m * 4096 + n] = tobf(v);
        }
      }
}

// ---------------------------------------------------------------------------
// Windowed attention: one block per (window, head). 8x8 window (64 pix),
// head dim 96. Phase1: S = (Q^T K)*scale via WMMA; Phase2: softmax (wave32
// shuffles); Phase3: O = V @ P^T via WMMA. Q,K,V,S,P all resident in LDS.
// ---------------------------------------------------------------------------
__global__ __launch_bounds__(256)
void attn_kernel(const __bf16* __restrict__ qkv, __bf16* __restrict__ outO) {
  __shared__ __bf16 QT[64][96];   // [d][c]
  __shared__ __bf16 KT[64][96];   // [e][c]
  __shared__ __bf16 Vl[96][64];   // [c][e]
  __shared__ float  Sf[64][64];   // scores [d][e]
  __shared__ __bf16 Pl[64][64];   // softmax(P)[d][e]
  const int win = blockIdx.x, head = blockIdx.y;
  const int wy = win >> 3, wx = win & 7;
  const int pbase = wy * 8 * 64 + wx * 8;
  const int tid = threadIdx.x, lane = tid & 31, wave = tid >> 5;

  for (int i = tid; i < 96 * 64; i += 256) {
    int c = i >> 6, d = i & 63;
    int p = pbase + (d >> 3) * 64 + (d & 7);
    QT[d][c] = qkv[(size_t)(head * 96 + c) * 4096 + p];
    KT[d][c] = qkv[(size_t)(384 + head * 96 + c) * 4096 + p];
    Vl[c][d] = qkv[(size_t)(768 + head * 96 + c) * 4096 + p];
  }
  __syncthreads();

  {  // Phase 1: 16 tiles of 16x16 over K=96, 2 per wave
    const int td = wave >> 1, te0 = (wave & 1) * 2;
    const int kb2 = (lane >> 4) * 8, kh = (lane >> 4) * 16;
    const int rowq = td * 16 + (lane & 15);
    for (int tc = te0; tc < te0 + 2; ++tc) {
      const int cole = tc * 16 + (lane & 15);
      v8f acc = {};
#pragma unroll
      for (int k0 = 0; k0 < 96; k0 += 32) {
        v16bf a = load_frag_a(&QT[rowq][k0], kb2);
        v16bf b = *(const v16bf*)(&KT[cole][k0 + kh]);
        acc = __builtin_amdgcn_wmma_f32_16x16x32_bf16(false, a, false, b, (short)0, acc, false, false);
      }
#pragma unroll
      for (int vg = 0; vg < 8; ++vg)
        Sf[td * 16 + vg + 8 * (lane >> 4)][tc * 16 + (lane & 15)] =
            acc[vg] * 0.10206207261596575f;  // (96)^-0.5
    }
  }
  __syncthreads();

  // Phase 2: per-row softmax, one row per wave (2 elems/lane), wave32 reduce
  for (int r = wave; r < 64; r += 8) {
    float a0 = Sf[r][lane], a1 = Sf[r][lane + 32];
    float mx = fmaxf(a0, a1);
#pragma unroll
    for (int off = 16; off > 0; off >>= 1) mx = fmaxf(mx, __shfl_xor(mx, off, 32));
    float e0 = __expf(a0 - mx), e1 = __expf(a1 - mx);
    float sm = e0 + e1;
#pragma unroll
    for (int off = 16; off > 0; off >>= 1) sm += __shfl_xor(sm, off, 32);
    float inv = 1.0f / sm;
    Pl[r][lane]      = tobf(e0 * inv);
    Pl[r][lane + 32] = tobf(e1 * inv);
  }
  __syncthreads();

  {  // Phase 3: O[c,d] = sum_e V[c,e] P[d,e] -> 24 tiles (96x64), 3 per wave
    const int kb2 = (lane >> 4) * 8, kh = (lane >> 4) * 16;
    for (int t = wave; t < 24; t += 8) {
      const int mt = t >> 2, nt = t & 3;
      const int rowc = mt * 16 + (lane & 15);
      const int cold = nt * 16 + (lane & 15);
      v8f acc = {};
#pragma unroll
      for (int k0 = 0; k0 < 64; k0 += 32) {
        v16bf a = load_frag_a(&Vl[rowc][k0], kb2);
        v16bf b = *(const v16bf*)(&Pl[cold][k0 + kh]);
        acc = __builtin_amdgcn_wmma_f32_16x16x32_bf16(false, a, false, b, (short)0, acc, false, false);
      }
#pragma unroll
      for (int vg = 0; vg < 8; ++vg) {
        int ch = head * 96 + mt * 16 + vg + 8 * (lane >> 4);
        int dc = nt * 16 + (lane & 15);
        int p = pbase + (dc >> 3) * 64 + (dc & 7);
        outO[(size_t)ch * 4096 + p] = tobf(acc[vg]);
      }
    }
  }
}

// ---------------------------------------------------------------------------
// Small VALU kernels
// ---------------------------------------------------------------------------
__global__ __launch_bounds__(256)
void cvt_pad_kernel(const float* __restrict__ src, __bf16* __restrict__ dst,
                    int M, int K, int Mp, int Kp) {
  int i = blockIdx.x * 256 + threadIdx.x;
  if (i >= Mp * Kp) return;
  int m = i / Kp, k = i - m * Kp;
  float v = (m < M && k < K) ? src[m * K + k] : 0.0f;
  dst[i] = tobf(v);
}

__global__ __launch_bounds__(256)
void bbcb_kernel(const float* __restrict__ Bf, const float* __restrict__ Bo,
                 const float* __restrict__ Cf, const float* __restrict__ Co,
                 float* __restrict__ BbCb) {
  int i = blockIdx.x * 256 + threadIdx.x;
  if (i >= 6144) return;
  int d = i >> 4, s = i & 15;
  float b = 0.f, c = 0.f;
#pragma unroll
  for (int r = 0; r < 12; ++r) {
    b += Bf[d * 12 + r] * Bo[r * 16 + s];
    c += Cf[d * 12 + r] * Co[r * 16 + s];
  }
  BbCb[i] = b; BbCb[6144 + i] = c;
}

__global__ __launch_bounds__(256)
void rmsnorm_bf16_kernel(const float* __restrict__ x, const float* __restrict__ w,
                         __bf16* __restrict__ out, int C) {
  int p = blockIdx.x * 256 + threadIdx.x;
  float ss = 0.f;
  for (int c = 0; c < C; ++c) { float v = x[(size_t)c * 4096 + p]; ss += v * v; }
  float rs = rsqrtf(ss / (float)C + 1e-5f);
  for (int c = 0; c < C; ++c)
    out[(size_t)c * 4096 + p] = tobf(x[(size_t)c * 4096 + p] * rs * w[c]);
}

__global__ __launch_bounds__(256)
void dwconv_silu_kernel(const float* __restrict__ xa, const float* __restrict__ dw_w,
                        const float* __restrict__ dw_b, float* __restrict__ t) {
  int p = blockIdx.x * 256 + threadIdx.x;
  int c = blockIdx.y;
  int py = p >> 6, px = p & 63;
  float acc = dw_b[c];
#pragma unroll
  for (int ky = 0; ky < 3; ++ky) {
    int iy = py + ky - 1;
    if (iy < 0 || iy > 63) continue;
#pragma unroll
    for (int kx = 0; kx < 3; ++kx) {
      int ix = px + kx - 1;
      if (ix < 0 || ix > 63) continue;
      acc += xa[(size_t)c * 4096 + iy * 64 + ix] * dw_w[c * 9 + ky * 3 + kx];
    }
  }
  t[(size_t)c * 4096 + p] = acc * sigf(acc);
}

__global__ __launch_bounds__(256)
void delta_kernel(const float* __restrict__ pbuf, const float* __restrict__ dlw,
                  const float* __restrict__ dlb, float* __restrict__ delta) {
  int p = blockIdx.x * 256 + threadIdx.x;
  int d = blockIdx.y;
  float a = dlb[d];
#pragma unroll
  for (int r = 0; r < 12; ++r) a += dlw[d * 12 + r] * pbuf[(size_t)r * 4096 + p];
  delta[(size_t)d * 4096 + p] = sigf(a);
}

__global__ __launch_bounds__(256)
void amodU_kernel(const float* __restrict__ xaF, const float* __restrict__ aw,
                  const float* __restrict__ ab, float* __restrict__ U) {
  int p = blockIdx.x * 256 + threadIdx.x;
  float u0 = ab[0], u1 = ab[1];
  for (int c = 0; c < 384; ++c) {
    float v = xaF[(size_t)c * 4096 + p];
    u0 += aw[c] * v;
    u1 += aw[384 + c] * v;
  }
  U[p] = u0; U[4096 + p] = u1;
}

// ---------------------------------------------------------------------------
// Fused SSM state update: the bandwidth-critical pass (206 MB h traffic).
// One thread per (d, p); 16 states in registers; coalesced along p.
// ---------------------------------------------------------------------------
__global__ __launch_bounds__(256)
void state_kernel(const float* __restrict__ delta, const float* __restrict__ xaF,
                  const float* __restrict__ gB, const float* __restrict__ gC,
                  const float* __restrict__ U, const float* __restrict__ pbuf,
                  const float* __restrict__ h_prev, const float* __restrict__ A_log,
                  const float* __restrict__ AmodV, const float* __restrict__ Bfac,
                  const float* __restrict__ Cfac, const float* __restrict__ Bout,
                  const float* __restrict__ Cout, const float* __restrict__ BbCb,
                  const float* __restrict__ Dskip, const float* __restrict__ xr,
                  const float* __restrict__ sc_amod, const float* __restrict__ sc_dyn,
                  const float* __restrict__ sc_ssm,
                  float* __restrict__ h_new, __bf16* __restrict__ yB) {
  __shared__ float BoL[192], CoL[192];
  int tid = threadIdx.x;
  if (tid < 192) { BoL[tid] = Bout[tid]; CoL[tid] = Cout[tid]; }
  __syncthreads();
  int p = blockIdx.x * 256 + tid;
  int d = blockIdx.y;

  float du = delta[(size_t)d * 4096 + p];
  float xa = xaF[(size_t)d * 4096 + p];
  float gb = gB[(size_t)d * 4096 + p];
  float gc = gC[(size_t)d * 4096 + p];
  float u0 = U[p], u1 = U[4096 + p];
  float wB[12], wC[12];
#pragma unroll
  for (int r = 0; r < 12; ++r) {
    wB[r] = pbuf[(size_t)(12 + r) * 4096 + p] * Bfac[d * 12 + r];
    wC[r] = pbuf[(size_t)(24 + r) * 4096 + p] * Cfac[d * 12 + r];
  }
  float amods = *sc_amod, dyn = *sc_dyn, ssm = *sc_ssm;
  float acc = 0.f;
#pragma unroll 4
  for (int s = 0; s < 16; ++s) {
    float bd = 0.f, cd = 0.f;
#pragma unroll
    for (int r = 0; r < 12; ++r) { bd += wB[r] * BoL[r * 16 + s]; cd += wC[r] * CoL[r * 16 + s]; }
    float nA = -__expf(A_log[d * 16 + s]);
    float am = u0 * AmodV[(size_t)d * 16 + s] + u1 * AmodV[(size_t)6144 + d * 16 + s];
    float A = nA * (1.f + amods * tanhf(am));
    float Beff = (BbCb[d * 16 + s] + dyn * tanhf(bd)) * gb;
    float Ceff = (BbCb[6144 + d * 16 + s] + dyn * tanhf(cd)) * gc;
    size_t hidx = (size_t)(d * 16 + s) * 4096 + p;
    float hn = __expf(A * du) * h_prev[hidx] + du * xa * Beff;
    h_new[hidx] = hn;
    acc += hn * Ceff;
  }
  float y = acc + xa * Dskip[d];
  float rv = xr[(size_t)(384 + d) * 4096 + p];  // res half of in_proj output
  yB[(size_t)d * 4096 + p] = tobf(ssm * y * (rv * sigf(rv)));
}

// ---------------------------------------------------------------------------
// Host orchestration
// ---------------------------------------------------------------------------
static inline int cdiv(int a, int b) { return (a + b - 1) / b; }

extern "C" void kernel_launch(void* const* d_in, const int* in_sizes, int n_in,
                              void* d_out, int out_size, void* d_ws, size_t ws_size,
                              hipStream_t stream) {
  (void)in_sizes; (void)n_in; (void)out_size; (void)ws_size;
  const float* x          = (const float*)d_in[0];
  const float* h_prev     = (const float*)d_in[1];
  const float* norm_w     = (const float*)d_in[2];
  const float* attn_nw    = (const float*)d_in[3];
  const float* in_proj_w  = (const float*)d_in[4];
  const float* in_proj_b  = (const float*)d_in[5];
  const float* dw_w       = (const float*)d_in[6];
  const float* dw_b       = (const float*)d_in[7];
  const float* qkv_w      = (const float*)d_in[8];
  const float* attn_out_w = (const float*)d_in[9];
  const float* attn_out_b = (const float*)d_in[10];
  const float* xproj_w    = (const float*)d_in[11];
  const float* xproj_b    = (const float*)d_in[12];
  const float* dlift_w    = (const float*)d_in[13];
  const float* dlift_b    = (const float*)d_in[14];
  const float* gateB_w    = (const float*)d_in[15];
  const float* gateB_b    = (const float*)d_in[16];
  const float* gateC_w    = (const float*)d_in[17];
  const float* gateC_b    = (const float*)d_in[18];
  const float* B_factor   = (const float*)d_in[19];
  const float* B_out      = (const float*)d_in[20];
  const float* C_factor   = (const float*)d_in[21];
  const float* C_out      = (const float*)d_in[22];
  const float* A_log      = (const float*)d_in[23];
  const float* Amod_w     = (const float*)d_in[24];
  const float* Amod_b     = (const float*)d_in[25];
  const float* Amod_V     = (const float*)d_in[26];
  const float* Amod_scale = (const float*)d_in[27];
  const float* D_skip     = (const float*)d_in[28];
  const float* dyn_scale  = (const float*)d_in[29];
  const float* out_w      = (const float*)d_in[30];
  const float* out_b      = (const float*)d_in[31];
  const float* ssm_scale  = (const float*)d_in[32];

  const int P = 4096;
  float* outP  = (float*)d_out;
  float* h_new = outP + (size_t)192 * P;

  // ---- workspace layout (~55 MB; all buffers written before read) ----
  uint8_t* w8 = (uint8_t*)d_ws;
  size_t off = 0;
  auto take = [&](size_t bytes) -> void* {
    void* p = w8 + off;
    off = (off + bytes + 255) & ~(size_t)255;
    return p;
  };
  __bf16* wb_inproj  = (__bf16*)take((size_t)768 * 192 * 2);
  __bf16* wb_qkv     = (__bf16*)take((size_t)1152 * 384 * 2);
  __bf16* wb_attnout = (__bf16*)take((size_t)384 * 384 * 2);
  __bf16* wb_gateB   = (__bf16*)take((size_t)384 * 384 * 2);
  __bf16* wb_gateC   = (__bf16*)take((size_t)384 * 384 * 2);
  __bf16* wb_xproj   = (__bf16*)take((size_t)64 * 384 * 2);   // padded 36->64 rows
  __bf16* wb_out     = (__bf16*)take((size_t)192 * 384 * 2);
  __bf16* xn_b  = (__bf16*)take((size_t)192 * P * 2);
  float*  xr    = (float*) take((size_t)768 * P * 4);   // [xa_raw | res]
  float*  t     = (float*) take((size_t)384 * P * 4);
  __bf16* xan_b = (__bf16*)take((size_t)384 * P * 2);
  __bf16* qkv_b = (__bf16*)take((size_t)1152 * P * 2);
  __bf16* attn_b= (__bf16*)take((size_t)384 * P * 2);
  float*  xaF   = (float*) take((size_t)384 * P * 4);
  __bf16* xaB   = (__bf16*)take((size_t)384 * P * 2);
  float*  pbuf  = (float*) take((size_t)36 * P * 4);
  float*  gC    = (float*) take((size_t)384 * P * 4);
  float*  U     = (float*) take((size_t)2 * P * 4);
  float*  bbcb  = (float*) take((size_t)2 * 6144 * 4);
  // dead-buffer aliases (t dead after rmsnorm2; qkv_b dead after attention;
  // attn_b dead after attn_out GEMM)
  float*  delta = t;
  float*  gB    = (float*)qkv_b;
  __bf16* yB    = attn_b;

  // ---- weight conversion (f32 -> bf16, zero-padded tiles) ----
  cvt_pad_kernel<<<cdiv(768 * 192, 256), 256, 0, stream>>>(in_proj_w, wb_inproj, 768, 192, 768, 192);
  cvt_pad_kernel<<<cdiv(1152 * 384, 256), 256, 0, stream>>>(qkv_w, wb_qkv, 1152, 384, 1152, 384);
  cvt_pad_kernel<<<cdiv(384 * 384, 256), 256, 0, stream>>>(attn_out_w, wb_attnout, 384, 384, 384, 384);
  cvt_pad_kernel<<<cdiv(384 * 384, 256), 256, 0, stream>>>(gateB_w, wb_gateB, 384, 384, 384, 384);
  cvt_pad_kernel<<<cdiv(384 * 384, 256), 256, 0, stream>>>(gateC_w, wb_gateC, 384, 384, 384, 384);
  cvt_pad_kernel<<<cdiv(64 * 384, 256), 256, 0, stream>>>(xproj_w, wb_xproj, 36, 384, 64, 384);
  cvt_pad_kernel<<<cdiv(192 * 384, 256), 256, 0, stream>>>(out_w, wb_out, 192, 384, 192, 384);
  bbcb_kernel<<<24, 256, 0, stream>>>(B_factor, B_out, C_factor, C_out, bbcb);

  // ---- forward pipeline ----
  rmsnorm_bf16_kernel<<<16, 256, 0, stream>>>(x, norm_w, xn_b, 192);
  gemm_wmma_kernel<0><<<dim3(32, 12), 256, 0, stream>>>(wb_inproj, xn_b, in_proj_b,
      nullptr, xr, nullptr, 768, 192);
  dwconv_silu_kernel<<<dim3(16, 384), 256, 0, stream>>>(xr, dw_w, dw_b, t);
  rmsnorm_bf16_kernel<<<16, 256, 0, stream>>>(t, attn_nw, xan_b, 384);
  gemm_wmma_kernel<0><<<dim3(32, 18), 256, 0, stream>>>(wb_qkv, xan_b, nullptr,
      nullptr, nullptr, qkv_b, 1152, 384);
  attn_kernel<<<dim3(64, 4), 256, 0, stream>>>(qkv_b, attn_b);
  gemm_wmma_kernel<0><<<dim3(32, 6), 256, 0, stream>>>(wb_attnout, attn_b, attn_out_b,
      nullptr, xaF, xaB, 384, 384);
  gemm_wmma_kernel<0><<<dim3(32, 1), 256, 0, stream>>>(wb_xproj, xaB, xproj_b,
      nullptr, pbuf, nullptr, 36, 384);
  delta_kernel<<<dim3(16, 384), 256, 0, stream>>>(pbuf, dlift_w, dlift_b, delta);
  gemm_wmma_kernel<1><<<dim3(32, 6), 256, 0, stream>>>(wb_gateB, xaB, gateB_b,
      nullptr, gB, nullptr, 384, 384);
  gemm_wmma_kernel<1><<<dim3(32, 6), 256, 0, stream>>>(wb_gateC, xaB, gateC_b,
      nullptr, gC, nullptr, 384, 384);
  amodU_kernel<<<16, 256, 0, stream>>>(xaF, Amod_w, Amod_b, U);
  state_kernel<<<dim3(16, 384), 256, 0, stream>>>(delta, xaF, gB, gC, U, pbuf,
      h_prev, A_log, Amod_V, B_factor, C_factor, B_out, C_out, bbcb, D_skip, xr,
      Amod_scale, dyn_scale, ssm_scale, h_new, yB);
  gemm_wmma_kernel<0><<<dim3(32, 3), 256, 0, stream>>>(wb_out, yB, out_b,
      x, outP, nullptr, 192, 384);
}
